// SwinTransformerBlock_71305047048375
// MI455X (gfx1250) — matrix operand order repaired and checked by reading, not compiled, not run on hardware
//
#include <hip/hip_runtime.h>
#include <hip/hip_bf16.h>
#include <math.h>
#include <stdint.h>

// ---------------- types ----------------
typedef __attribute__((ext_vector_type(8)))  float  v8f;
typedef __attribute__((ext_vector_type(4)))  float  v4f;
typedef __attribute__((ext_vector_type(16))) __bf16 v16bf;
typedef __attribute__((ext_vector_type(8)))  __bf16 v8bf;
typedef __attribute__((ext_vector_type(4)))  __bf16 v4bf;

#define SS_    4
#define MROWS  262144      // B * H * W = 16 * 128 * 128 tokens

// ---------------- WMMA helpers ----------------
__device__ __forceinline__ v8f zero8() {
  v8f z;
#pragma unroll
  for (int i = 0; i < 8; ++i) z[i] = 0.0f;
  return z;
}

__device__ __forceinline__ v8f wmma_bf16(v16bf a, v16bf b, v8f c) {
  // D = A(16xK=32, bf16) * B(32x16, bf16) + C(f32)
  return __builtin_amdgcn_wmma_f32_16x16x32_bf16(false, a, false, b, (short)0, c,
                                                 false, false);
}

// A fragment: lane (r = lane&15, h = lane>>4) holds row (row0+r),
// elements {k0+8h .. k0+8h+7} then {k0+16+8h .. +7}  (ISA 16-bit A 16x32 layout)
__device__ __forceinline__ v16bf load_frag_a(const __bf16* base, int ld, int row0,
                                             int k0, int lane) {
  int r = lane & 15, h = lane >> 4;
  const __bf16* p = base + (size_t)(row0 + r) * ld + k0 + h * 8;
  v8bf a0 = *(const v8bf*)(p);
  v8bf a1 = *(const v8bf*)(p + 16);
  return __builtin_shufflevector(a0, a1, 0, 1, 2, 3, 4, 5, 6, 7, 8, 9, 10, 11, 12,
                                 13, 14, 15);
}

// B fragment (W is NxK row-major, i.e. B^T): lane (r,h) holds column (col0+r),
// elements {k0+8h .. k0+8h+7} then {k0+16+8h .. +7}  (same k order as A)
__device__ __forceinline__ v16bf load_frag_b(const __bf16* base, int ld, int col0,
                                             int k0, int lane) {
  int r = lane & 15, h = lane >> 4;
  const __bf16* p = base + (size_t)(col0 + r) * ld + k0 + h * 8;
  v8bf b0 = *(const v8bf*)(p);
  v8bf b1 = *(const v8bf*)(p + 16);
  return __builtin_shufflevector(b0, b1, 0, 1, 2, 3, 4, 5, 6, 7, 8, 9, 10, 11, 12,
                                 13, 14, 15);
}

// ---------------- fp32 -> bf16 weight convert ----------------
__global__ void cvt_bf16_kernel(const float* __restrict__ src,
                                __bf16* __restrict__ dst, int n) {
  int i = blockIdx.x * 256 + threadIdx.x;
  if (i < n) dst[i] = (__bf16)src[i];
}

// ---------------- LayerNorm (optionally fused shift + window partition) ----------
// One wave per token row; lane holds 4 channels of the 128.
template <bool PERM>
__global__ __launch_bounds__(256) void ln_kernel(const float* __restrict__ x,
                                                 const float* __restrict__ g,
                                                 const float* __restrict__ bta,
                                                 __bf16* __restrict__ out) {
  int lane = threadIdx.x & 31;
  int r = blockIdx.x * 8 + (threadIdx.x >> 5);  // output row (window space if PERM)
  int src = r;
  if (PERM) {
    int b   = r >> 14;          // image
    int rem = r & 16383;
    int wh  = rem >> 10;        // window row  (16 per image)
    int ww  = (rem >> 6) & 15;  // window col
    int n   = rem & 63;         // token in window
    int i = n >> 3, j = n & 7;
    int h = (wh * 8 + i + SS_) & 127;  // cyclic shift by -SS == gather at +SS
    int w = (ww * 8 + j + SS_) & 127;
    src = b * 16384 + h * 128 + w;
  }
  v4f v = *(const v4f*)(x + (size_t)src * 128 + lane * 4);
  float s  = v.x + v.y + v.z + v.w;
  float s2 = v.x * v.x + v.y * v.y + v.z * v.z + v.w * v.w;
#pragma unroll
  for (int off = 16; off; off >>= 1) {
    s  += __shfl_xor(s, off, 32);
    s2 += __shfl_xor(s2, off, 32);
  }
  float m    = s * (1.0f / 128.0f);
  float rstd = rsqrtf(s2 * (1.0f / 128.0f) - m * m + 1e-5f);
  v4bf o;
#pragma unroll
  for (int q = 0; q < 4; ++q) {
    int c = lane * 4 + q;
    o[q] = (__bf16)((v[q] - m) * rstd * g[c] + bta[c]);
  }
  *(v4bf*)(out + (size_t)r * 128 + lane * 4) = o;
}

// ---------------- generic bf16 GEMM  D = A * W^T + bias, fused epilogues --------
// MODE 0: store bf16 (QKV)          MODE 2: GELU(exact) -> bf16 (FC1)
// MODE 1: window-reverse+unshift, + residual -> f32 (proj)
// MODE 3: + residual -> f32 (FC2, final output)
template <int MODE>
__global__ __launch_bounds__(256) void gemm_bf16_kernel(
    const __bf16* __restrict__ A, const __bf16* __restrict__ W,
    const float* __restrict__ bias, int K, int N, __bf16* __restrict__ outB,
    float* __restrict__ outF, const float* __restrict__ resid) {
  int lane = threadIdx.x & 31;
  int wave = threadIdx.x >> 5;
  int row0 = (blockIdx.x * 8 + wave) * 16;  // 16-row strip per wave
  int n0   = blockIdx.y * 64;               // 64-col group per wave (4 tiles)

  v8f acc[4];
#pragma unroll
  for (int t = 0; t < 4; ++t) acc[t] = zero8();

  for (int k0 = 0; k0 < K; k0 += 32) {
    v16bf a = load_frag_a(A, K, row0, k0, lane);
    if (k0 + 32 < K)  // hint the next A chunk (global_prefetch_b8)
      __builtin_prefetch(A + (size_t)(row0 + (lane & 15)) * K + k0 + 32, 0, 0);
#pragma unroll
    for (int t = 0; t < 4; ++t) {
      v16bf b = load_frag_b(W, K, n0 + t * 16, k0, lane);
      acc[t] = wmma_bf16(a, b, acc[t]);
    }
  }

  int cb = lane & 15, hf = lane >> 4;
#pragma unroll
  for (int t = 0; t < 4; ++t) {
    int col  = n0 + t * 16 + cb;
    float bv = bias[col];
#pragma unroll
    for (int j = 0; j < 8; ++j) {
      int row = row0 + j + 8 * hf;
      float v = acc[t][j] + bv;
      if (MODE == 0) {
        outB[(size_t)row * N + col] = (__bf16)v;
      } else if (MODE == 2) {
        float ge = 0.5f * v * (1.0f + erff(v * 0.70710678f));
        outB[(size_t)row * N + col] = (__bf16)ge;
      } else if (MODE == 3) {
        size_t o = (size_t)row * N + col;
        outF[o] = v + resid[o];
      } else {  // MODE 1: proj, window-reverse + reverse cyclic shift + residual
        int b_ = row >> 6, n = row & 63;
        int bimg = b_ >> 8, widx = b_ & 255;
        int wh = widx >> 4, ww = widx & 15;
        int i = n >> 3, jj = n & 7;
        int h = (wh * 8 + i + SS_) & 127;   // roll(+SS): final[(h'+SS)%H] = t[h']
        int w = (ww * 8 + jj + SS_) & 127;
        size_t pos = (size_t)bimg * 16384 + (size_t)h * 128 + w;
        size_t o   = pos * 128 + col;
        outF[o] = v + resid[o];
      }
    }
  }
}

// ---------------- windowed attention: one block per (window, head) --------------
// 4 waves: wave w owns rows 16w..16w+15.  hd = 32 == one bf16 WMMA K-step.
__global__ __launch_bounds__(128) void attn_kernel(const __bf16* __restrict__ QKV,
                                                   const float* __restrict__ rpb,
                                                   const int* __restrict__ relidx,
                                                   const float* __restrict__ mask,
                                                   __bf16* __restrict__ AO) {
  __shared__ __align__(16) float  sS[64 * 64];   // logits / probs (f32)
  __shared__ __align__(16) __bf16 sP[64 * 64];   // probs (bf16)
  __shared__ __align__(16) __bf16 sVt[32 * 64];  // V^T : [d][n]

  int b_ = blockIdx.x, head = blockIdx.y;
  int tid = threadIdx.x, lane = tid & 31, wave = tid >> 5;
  int cb = lane & 15, hf = lane >> 4;

  const __bf16* qb = QKV + (size_t)b_ * 64 * 384 + head * 32;
  const __bf16* kb = qb + 128;
  const __bf16* vb = qb + 256;

  // stage V^T into LDS
  for (int idx = tid; idx < 2048; idx += 128) {
    int n = idx >> 5, d = idx & 31;
    sVt[d * 64 + n] = vb[(size_t)n * 384 + d];
  }

  // S = (q + bq) (k + bk)^T * scale   (scale folded here; bias is in QKV)
  {
    v16bf a = load_frag_a(qb, 384, wave * 16, 0, lane);
#pragma unroll
    for (int t = 0; t < 4; ++t) {
      v16bf kf = load_frag_b(kb, 384, t * 16, 0, lane);
      v8f s = wmma_bf16(a, kf, zero8());
#pragma unroll
      for (int j = 0; j < 8; ++j)
        sS[(wave * 16 + j + 8 * hf) * 64 + t * 16 + cb] = s[j] * 0.17677669529f;
    }
  }
  __syncthreads();

  // softmax rows with relative-position bias + shift mask
  if (tid < 64) {
    int row  = tid;
    int widx = b_ & 255;
    const float* mrow = mask + ((size_t)widx * 64 + row) * 64;
    const int*   irow = relidx + row * 64;
    float mx = -1e30f;
    for (int j = 0; j < 64; ++j) {
      float t = sS[row * 64 + j] + rpb[irow[j] * 4 + head] + mrow[j];
      sS[row * 64 + j] = t;
      mx = fmaxf(mx, t);
    }
    float sum = 0.0f;
    for (int j = 0; j < 64; ++j) {
      float e = __expf(sS[row * 64 + j] - mx);
      sS[row * 64 + j] = e;
      sum += e;
    }
    float inv = 1.0f / sum;
    for (int j = 0; j < 64; ++j) sP[row * 64 + j] = (__bf16)(sS[row * 64 + j] * inv);
  }
  __syncthreads();

  // O = P V   (A from LDS sP, B from LDS sVt; K=64 -> 2 WMMA steps)
  v8f acc[2];
  acc[0] = zero8();
  acc[1] = zero8();
#pragma unroll
  for (int k0 = 0; k0 < 64; k0 += 32) {
    v16bf pa = load_frag_a(sP, 64, wave * 16, k0, lane);
#pragma unroll
    for (int t = 0; t < 2; ++t) {
      v16bf vf = load_frag_b(sVt, 64, t * 16, k0, lane);
      acc[t] = wmma_bf16(pa, vf, acc[t]);
    }
  }
#pragma unroll
  for (int t = 0; t < 2; ++t)
#pragma unroll
    for (int j = 0; j < 8; ++j) {
      int row = wave * 16 + j + 8 * hf;
      AO[((size_t)b_ * 64 + row) * 128 + head * 32 + t * 16 + cb] =
          (__bf16)acc[t][j];
    }
}

// ---------------- launcher ----------------
extern "C" void kernel_launch(void* const* d_in, const int* in_sizes, int n_in,
                              void* d_out, int out_size, void* d_ws, size_t ws_size,
                              hipStream_t stream) {
  (void)in_sizes; (void)n_in; (void)out_size; (void)ws_size;
  const float* x      = (const float*)d_in[0];
  const float* g1     = (const float*)d_in[1];
  const float* b1     = (const float*)d_in[2];
  const float* qkv_w  = (const float*)d_in[3];
  const float* qkv_b  = (const float*)d_in[4];
  const float* proj_w = (const float*)d_in[5];
  const float* proj_b = (const float*)d_in[6];
  const float* rpb    = (const float*)d_in[7];
  const float* g2     = (const float*)d_in[8];
  const float* b2     = (const float*)d_in[9];
  const float* fc1_w  = (const float*)d_in[10];
  const float* fc1_b  = (const float*)d_in[11];
  const float* fc2_w  = (const float*)d_in[12];
  const float* fc2_b  = (const float*)d_in[13];
  const int*   relidx = (const int*)d_in[14];
  const float* amask  = (const float*)d_in[15];

  // workspace carve-up (regions reused across pipeline stages)
  char* p = (char*)d_ws;
  __bf16* Wq  = (__bf16*)p; p += (size_t)384 * 128 * 2;
  __bf16* Wp  = (__bf16*)p; p += (size_t)128 * 128 * 2;
  __bf16* Wf1 = (__bf16*)p; p += (size_t)512 * 128 * 2;
  __bf16* Wf2 = (__bf16*)p; p += (size_t)128 * 512 * 2;
  p = (char*)(((uintptr_t)p + 255) & ~(uintptr_t)255);
  __bf16* RA = (__bf16*)p; p += (size_t)MROWS * 128 * 2;  //  64MB: Hln -> AO -> H2
  __bf16* RB = (__bf16*)p; p += (size_t)MROWS * 512 * 2;  // 256MB: QKV -> HMLP
  float*  X1 = (float*)p;  p += (size_t)MROWS * 128 * 4;  // 128MB: post-attn resid

  // 1) weights -> bf16 (tiny; L2-resident thereafter)
  cvt_bf16_kernel<<<(384 * 128 + 255) / 256, 256, 0, stream>>>(qkv_w, Wq, 384 * 128);
  cvt_bf16_kernel<<<(128 * 128 + 255) / 256, 256, 0, stream>>>(proj_w, Wp, 128 * 128);
  cvt_bf16_kernel<<<(512 * 128 + 255) / 256, 256, 0, stream>>>(fc1_w, Wf1, 512 * 128);
  cvt_bf16_kernel<<<(128 * 512 + 255) / 256, 256, 0, stream>>>(fc2_w, Wf2, 128 * 512);

  // 2) LN1 fused with cyclic shift + window partition -> RA (bf16)
  ln_kernel<true><<<MROWS / 8, 256, 0, stream>>>(x, g1, b1, RA);

  // 3) QKV GEMM: (262144x128) x (384x128)^T -> RB (bf16)
  gemm_bf16_kernel<0><<<dim3(MROWS / 128, 6), 256, 0, stream>>>(
      RA, Wq, qkv_b, 128, 384, RB, nullptr, nullptr);

  // 4) windowed MHA per (window, head) -> RA (bf16, heads merged back to C=128)
  attn_kernel<<<dim3(4096, 4), 128, 0, stream>>>(RB, rpb, relidx, amask, RA);

  // 5) proj + window-reverse + unshift + residual -> X1 (f32)
  gemm_bf16_kernel<1><<<dim3(MROWS / 128, 2), 256, 0, stream>>>(
      RA, Wp, proj_b, 128, 128, nullptr, X1, x);

  // 6) LN2 -> RA (bf16)
  ln_kernel<false><<<MROWS / 8, 256, 0, stream>>>(X1, g2, b2, RA);

  // 7) FC1 + exact GELU -> RB (bf16, 512 wide)
  gemm_bf16_kernel<2><<<dim3(MROWS / 128, 8), 256, 0, stream>>>(
      RA, Wf1, fc1_b, 128, 512, RB, nullptr, nullptr);

  // 8) FC2 + residual -> d_out (f32)
  gemm_bf16_kernel<3><<<dim3(MROWS / 128, 2), 256, 0, stream>>>(
      RB, Wf2, fc2_b, 512, 128, nullptr, (float*)d_out, X1);
}
